// INGPNetwork_48782238548485
// MI455X (gfx1250) — matrix operand order repaired and checked by reading, not compiled, not run on hardware
//
#include <hip/hip_runtime.h>

// ---------------- CDNA5 vector types ----------------
typedef __attribute__((ext_vector_type(16))) _Float16 v16h;
typedef __attribute__((ext_vector_type(8)))  _Float16 v8h;
typedef __attribute__((ext_vector_type(8)))  float    v8f;
typedef __attribute__((ext_vector_type(4)))  int      v4i;

#if __has_builtin(__builtin_amdgcn_global_load_async_to_lds_b128)
#define USE_ASYNC_STAGE 1
typedef __attribute__((address_space(1))) v4i gv4i;   // global int4
typedef __attribute__((address_space(3))) v4i lv4i;   // LDS int4
#endif

// ---------------- static hash-grid configuration (compile-time folded) ----------------
#define LVLS 16
__device__ constexpr int      k_res[LVLS]  = {16,23,31,43,59,81,112,154,213,295,407,562,777,1073,1483,2048};
__device__ constexpr unsigned k_off[LVLS]  = {0u,4920u,18744u,51512u,136696u,352696u,876984u,1401272u,
                                              1925560u,2449848u,2974136u,3498424u,4022712u,4547000u,
                                              5071288u,5595576u};
__device__ constexpr int      k_hash[LVLS] = {0,0,0,0,0,1,1,1,1,1,1,1,1,1,1,1};
#define HASH_MASK 0x7FFFFu          // all hashed levels have size 2^19
#define P1 2654435761u
#define P2 805459861u

// ---------------- staged blob geometry ----------------
// B-fragment layout: fragment (n,k) = 32 lanes x 16 halves contiguous (512 halves).
// lane L holds column (L&15), K = 32*k + (L>>4)*16 + j, j = 0..15.
#define WB_L0 0            // 32x128  : 8 tiles x 1 kfrag  = 4096 halves
#define WB_L1 4096         // 128x128 : 8 tiles x 4 kfrags = 16384
#define WB_L2 20480
#define WB_L3 36864
#define WB_L4 53248        // 128x16 (padded from 128x1): 4 frags = 2048
#define WB_TOT 55296       // halves
#define BIAS_F 516         // floats: b0..b3 [4][128] + b4 + 3 pad
#define STAGE_BYTES (WB_TOT * 2 + BIAS_F * 4)   // 112656, multiple of 16
#define STAGE_CHUNKS (STAGE_BYTES / 16)         // 7041 x 16B

#define ACT_HALVES  (32 * 128)                  // one 32x128 f16 buffer
#define ACT_PER_WAVE (2 * ACT_HALVES)           // double-buffered: 8192 halves
#define SMEM_BYTES (STAGE_BYTES + 8 * ACT_PER_WAVE * 2)  // 112656 + 131072 = 243728 B

// ---------------- prep: swizzle weights f32 -> f16 fragments, append biases ----------------
__global__ void ingp_prep_weights(const float* __restrict__ W0, const float* __restrict__ W1,
                                  const float* __restrict__ W2, const float* __restrict__ W3,
                                  const float* __restrict__ W4,
                                  const float* __restrict__ b0, const float* __restrict__ b1,
                                  const float* __restrict__ b2, const float* __restrict__ b3,
                                  const float* __restrict__ b4,
                                  _Float16* __restrict__ ws) {
  int idx = blockIdx.x * 256 + threadIdx.x;
  if (idx >= WB_TOT + BIAS_F) return;
  if (idx >= WB_TOT) {                      // bias region (f32)
    int j = idx - WB_TOT;
    float* fb = (float*)(ws + WB_TOT);
    float v;
    if (j < 128)       v = b0[j];
    else if (j < 256)  v = b1[j - 128];
    else if (j < 384)  v = b2[j - 256];
    else if (j < 512)  v = b3[j - 384];
    else if (j == 512) v = b4[0];
    else               v = 0.0f;
    fb[j] = v;
    return;
  }
  int li, base, kf;
  if      (idx < WB_L1) { li = 0; base = WB_L0; kf = 1; }
  else if (idx < WB_L2) { li = 1; base = WB_L1; kf = 4; }
  else if (idx < WB_L3) { li = 2; base = WB_L2; kf = 4; }
  else if (idx < WB_L4) { li = 3; base = WB_L3; kf = 4; }
  else                  { li = 4; base = WB_L4; kf = 4; }
  int rel  = idx - base;
  int frag = rel >> 9;
  int lane = (rel >> 4) & 31;
  int j    = rel & 15;
  int n    = frag / kf;
  int k    = frag - n * kf;
  int col  = 16 * n + (lane & 15);
  int row  = 32 * k + ((lane >> 4) << 4) + j;   // K index
  float v;
  if      (li == 0) v = W0[row * 128 + col];
  else if (li == 1) v = W1[row * 128 + col];
  else if (li == 2) v = W2[row * 128 + col];
  else if (li == 3) v = W3[row * 128 + col];
  else              v = (col == 0) ? W4[row] : 0.0f;   // pad N=1 -> 16
  ws[idx] = (_Float16)v;
}

// ---------------- fragment loaders ----------------
__device__ __forceinline__ v16h load_a(const _Float16* buf, int stride, int kfrag, int lane) {
  int row = lane & 15;
  int kb  = 32 * kfrag + ((lane >> 4) << 4);
  const _Float16* p = buf + row * stride + kb;
  v8h lo = *(const v8h*)(p);
  v8h hi = *(const v8h*)(p + 8);
  return __builtin_shufflevector(lo, hi, 0,1,2,3,4,5,6,7,8,9,10,11,12,13,14,15);
}

__device__ __forceinline__ v16h load_b(const _Float16* wbase, int frag, int lane) {
  return *(const v16h*)(wbase + frag * 512 + lane * 16);   // 32B aligned
}

// ---------------- one 32-row layer: in(32xK f16) -> out(32x128 f16), bias+ReLU ----------------
template <int KFRAGS, int STRIDE>
__device__ __forceinline__ void layer_fwd(const _Float16* inBuf, _Float16* outBuf,
                                          const _Float16* wBase, const float* biasRow,
                                          int lane) {
  v16h a0[KFRAGS], a1[KFRAGS];
#pragma unroll
  for (int k = 0; k < KFRAGS; ++k) {
    a0[k] = load_a(inBuf, STRIDE, k, lane);
    a1[k] = load_a(inBuf + 16 * STRIDE, STRIDE, k, lane);
  }
  const int coloff = lane & 15;
  const int rowoff = (lane >> 4) << 3;
#pragma unroll
  for (int n = 0; n < 8; ++n) {
    v8f acc0 = {}, acc1 = {};
#pragma unroll
    for (int k = 0; k < KFRAGS; ++k) {
      v16h b = load_b(wBase, n * KFRAGS + k, lane);   // one B feeds two WMMAs
      acc0 = __builtin_amdgcn_wmma_f32_16x16x32_f16(false, a0[k], false, b,
                                                    (short)0, acc0, false, false);
      acc1 = __builtin_amdgcn_wmma_f32_16x16x32_f16(false, a1[k], false, b,
                                                    (short)0, acc1, false, false);
    }
    float bv = biasRow[16 * n + coloff];
#pragma unroll
    for (int r = 0; r < 8; ++r) {
      float v0 = fmaxf(acc0[r] + bv, 0.0f);
      float v1 = fmaxf(acc1[r] + bv, 0.0f);
      outBuf[(rowoff + r) * 128 + 16 * n + coloff]        = (_Float16)v0;
      outBuf[(16 + rowoff + r) * 128 + 16 * n + coloff]   = (_Float16)v1;
    }
  }
}

// ---------------- fused encode + MLP: 256 pts/block, 32 pts/wave ----------------
__global__ void ingp_fused(const float* __restrict__ x, const float* __restrict__ emb,
                           const _Float16* __restrict__ ws,
                           float* __restrict__ out, int npts) {
  extern __shared__ char smemc[];
  _Float16*    ldsW    = (_Float16*)smemc;                    // weight fragments
  const float* ldsBias = (const float*)(smemc + WB_TOT * 2);  // [4][128] + b4
  _Float16*    acts    = (_Float16*)(smemc + STAGE_BYTES);    // 8 waves x 2 x (32x128)

  const int t    = threadIdx.x;
  const int lane = t & 31;
  const int wave = t >> 5;

  // ---- stage weight+bias blob into LDS ----
#ifdef USE_ASYNC_STAGE
  {
    const char* src = (const char*)ws;
    for (int i = t; i < STAGE_CHUNKS; i += 256) {
      // flat LDS address: low 32 bits are the LDS byte offset (ISA 10.2 aperture rules)
      unsigned loff = (unsigned)(unsigned long long)(smemc + 16 * i);
      unsigned long long gaddr = (unsigned long long)(src + 16 * i);
      __builtin_amdgcn_global_load_async_to_lds_b128((gv4i*)gaddr, (lv4i*)loff, 0, 0);
    }
  }
#else
  {
    const int4* src = (const int4*)ws;
    int4* dst = (int4*)smemc;
    for (int i = t; i < STAGE_CHUNKS; i += 256) dst[i] = src[i];
  }
#endif

  // ---- hash-grid encoding: thread t -> point t, all 16 levels ----
  {
    const int gp = blockIdx.x * 256 + t;
    if (gp < npts) {
      const float X = x[3 * gp + 0];
      const float Y = x[3 * gp + 1];
      const float Z = x[3 * gp + 2];
      _Float16* hrow = acts + wave * ACT_PER_WAVE + (t & 31) * 32;  // buf0, stride 32
#pragma unroll
      for (int l = 0; l < LVLS; ++l) {
        const float rm1 = (float)(k_res[l] - 1);
        float px = X * rm1 + 0.5f, py = Y * rm1 + 0.5f, pz = Z * rm1 + 0.5f;
        float fx = floorf(px), fy = floorf(py), fz = floorf(pz);
        float tx = px - fx, ty = py - fy, tz = pz - fz;
        unsigned ux = (unsigned)fx, uy = (unsigned)fy, uz = (unsigned)fz;
        const unsigned s   = (unsigned)(k_res[l] + 1);
        const unsigned off = k_off[l];
        float a0 = 0.0f, a1 = 0.0f;
#pragma unroll
        for (int c = 0; c < 8; ++c) {
          unsigned cx = ux + ((c >> 2) & 1);
          unsigned cy = uy + ((c >> 1) & 1);
          unsigned cz = uz + (c & 1);
          unsigned idx;
          if (k_hash[l]) idx = (cx ^ (cy * P1) ^ (cz * P2)) & HASH_MASK;
          else           idx = cx + cy * s + cz * s * s;
          const float2 g = ((const float2*)emb)[off + idx];
          float w = (((c >> 2) & 1) ? tx : 1.0f - tx) *
                    (((c >> 1) & 1) ? ty : 1.0f - ty) *
                    (((c      ) & 1) ? tz : 1.0f - tz);
          a0 += w * g.x;
          a1 += w * g.y;
        }
        hrow[2 * l + 0] = (_Float16)a0;
        hrow[2 * l + 1] = (_Float16)a1;
      }
    }
  }

#ifdef USE_ASYNC_STAGE
#if __has_builtin(__builtin_amdgcn_s_wait_asynccnt)
  __builtin_amdgcn_s_wait_asynccnt(0);
#else
  asm volatile("s_wait_asynccnt 0x0" ::: "memory");
#endif
#endif
  __syncthreads();   // weights staged + features written

  // ---- per-wave MLP over its 32-point tile ----
  _Float16* buf0 = acts + wave * ACT_PER_WAVE;
  _Float16* buf1 = buf0 + ACT_HALVES;

  // Layer 0: 32 -> 128 (input stride 32), then 3x (128 -> 128)
  layer_fwd<1, 32>(buf0, buf1, ldsW + WB_L0, ldsBias + 0 * 128, lane);
  layer_fwd<4, 128>(buf1, buf0, ldsW + WB_L1, ldsBias + 1 * 128, lane);
  layer_fwd<4, 128>(buf0, buf1, ldsW + WB_L2, ldsBias + 2 * 128, lane);
  layer_fwd<4, 128>(buf1, buf0, ldsW + WB_L3, ldsBias + 3 * 128, lane);

  // Layer 4: 128 -> 1 (W4 padded to 16 cols; column 0 is the result)
  {
    v16h a0[4], a1[4];
#pragma unroll
    for (int k = 0; k < 4; ++k) {
      a0[k] = load_a(buf0, 128, k, lane);
      a1[k] = load_a(buf0 + 16 * 128, 128, k, lane);
    }
    v8f acc0 = {}, acc1 = {};
#pragma unroll
    for (int k = 0; k < 4; ++k) {
      v16h b = load_b(ldsW + WB_L4, k, lane);
      acc0 = __builtin_amdgcn_wmma_f32_16x16x32_f16(false, a0[k], false, b,
                                                    (short)0, acc0, false, false);
      acc1 = __builtin_amdgcn_wmma_f32_16x16x32_f16(false, a1[k], false, b,
                                                    (short)0, acc1, false, false);
    }
    if ((lane & 15) == 0) {
      const int rowoff = (lane >> 4) << 3;
      const int gbase  = blockIdx.x * 256 + wave * 32 + rowoff;
      const float bv   = ldsBias[512];
#pragma unroll
      for (int r = 0; r < 8; ++r) {
        int gi0 = gbase + r;
        int gi1 = gbase + 16 + r;
        if (gi0 < npts) out[gi0] = acc0[r] + bv;
        if (gi1 < npts) out[gi1] = acc1[r] + bv;
      }
    }
  }
}

// ---------------- host launch ----------------
extern "C" void kernel_launch(void* const* d_in, const int* in_sizes, int n_in,
                              void* d_out, int out_size, void* d_ws, size_t ws_size,
                              hipStream_t stream) {
  const float* x   = (const float*)d_in[0];
  const float* emb = (const float*)d_in[1];
  const float* W0  = (const float*)d_in[2];
  const float* b0  = (const float*)d_in[3];
  const float* W1  = (const float*)d_in[4];
  const float* b1  = (const float*)d_in[5];
  const float* W2  = (const float*)d_in[6];
  const float* b2  = (const float*)d_in[7];
  const float* W3  = (const float*)d_in[8];
  const float* b3  = (const float*)d_in[9];
  const float* W4  = (const float*)d_in[10];
  const float* b4  = (const float*)d_in[11];
  float* out = (float*)d_out;
  _Float16* wblob = (_Float16*)d_ws;

  const int npts = in_sizes[0] / 3;

  // 1) swizzle weights into CDNA5 B-fragment layout + bias table
  ingp_prep_weights<<<(WB_TOT + BIAS_F + 255) / 256, 256, 0, stream>>>(
      W0, W1, W2, W3, W4, b0, b1, b2, b3, b4, wblob);

  // 2) fused encode + MLP; 256 points/block, 8 waves, ~238 KB dynamic LDS
  (void)hipFuncSetAttribute((const void*)ingp_fused,
                            hipFuncAttributeMaxDynamicSharedMemorySize, SMEM_BYTES);
  const int nblk = (npts + 255) / 256;
  ingp_fused<<<nblk, 256, SMEM_BYTES, stream>>>(x, emb, wblob, out, npts);
}
